// SupportFeaturePurifier_25769804424
// MI455X (gfx1250) — matrix operand order repaired and checked by reading, not compiled, not run on hardware
//
#include <hip/hip_runtime.h>
#include <hip/hip_bf16.h>
#include <math.h>

typedef __attribute__((ext_vector_type(16))) _Float16     v16h;
typedef __attribute__((ext_vector_type(8)))  _Float16     v8h;
typedef __attribute__((ext_vector_type(8)))  float        v8f;
typedef __attribute__((ext_vector_type(4)))  float        v4f;
typedef __attribute__((ext_vector_type(4)))  unsigned int v4u;
typedef __attribute__((ext_vector_type(8)))  int          v8i;
typedef __attribute__((ext_vector_type(4)))  int          v4i;

#define BATCH 8
#define CCH   512
#define HW    4096   // 64*64
#define EPSV  1e-8f

// ---------------------------------------------------------------------------
// TDM: DMA a 128(row) x 32(col) f16 tile (row stride 512 halves) from global
// into LDS, hardware-padding each 32-half row by 8 halves -> LDS pitch 40.
// D# layout per CDNA5 ISA ch.8: group0 = {count, lds_addr, global_addr, type},
// group1 = {flags/data_size/pad, tensor dims, tile dims, strides}.
// ---------------------------------------------------------------------------
static __device__ inline void tdm_load_a_tile(const _Float16* gptr, _Float16* ldsptr) {
    unsigned long long ga = (unsigned long long)(size_t)gptr;
    unsigned int       la = (unsigned int)(size_t)ldsptr;   // generic low 32b = LDS offset
    v4u g0 = { 1u,                                          // count=1, user mode
               la,                                          // lds_addr
               (unsigned int)(ga & 0xFFFFFFFFu),            // global_addr[31:0]
               ((unsigned int)((ga >> 32) & 0x01FFFFFFu))   // global_addr[56:32]
                   | 0x80000000u };                         // type=2 ("image")
    v8i g1 = { (int)((1u << 16)     // data_size = 1 -> 2 bytes
                   | (1u << 20)     // pad_enable
                   | (3u << 22)     // pad_interval code 3 = 16 dwords (one 32-half row)
                   | (3u << 25)),   // pad_amount  code 3 = 4 dwords (8 halves)
               (int)(512u << 16),   // tensor_dim0[15:0]=512 in bits[63:48]
               (int)(512u << 16),   // tensor_dim0[31:16]=0, tensor_dim1[15:0]=512
               (int)(32u  << 16),   // tensor_dim1[31:16]=0, tile_dim0=32
               (int)128,            // tile_dim1=128, tile_dim2=0
               (int)512,            // tensor_dim0_stride[31:0]=512
               0, 0 };              // stride hi / dim1_stride unused (2D tile)
    v4i z4 = {0, 0, 0, 0};
#if __has_include(<hip/amd_detail/amd_gfx1250_TDM.h>)
    v8i z8 = {0, 0, 0, 0, 0, 0, 0, 0};
    __builtin_amdgcn_tensor_load_to_lds(g0, g1, z4, z4, z8, 0);
#else
    __builtin_amdgcn_tensor_load_to_lds(g0, g1, z4, z4, 0);
#endif
}

// ---------------------------------------------------------------------------
// Kernel 1: mask downsample (nearest x4), 3x3 dilation, bg ring, mask sums,
// exact 2-pass EDT + normalize. One block per batch image, all in LDS.
// ---------------------------------------------------------------------------
__global__ __launch_bounds__(256) void prep_kernel(
    const int* __restrict__ mask,        // [B,1,256,256]
    float* __restrict__ ms_g,            // [B,4096]
    float* __restrict__ mbg_g,           // [B,4096]
    float* __restrict__ dist_g,          // [B,4096]
    float* __restrict__ sums_g)          // [0..7]=sum_ms, [8..15]=sum_mbg
{
    __shared__ float ms[4096];
    __shared__ float gg[4096];
    __shared__ float dd[4096];
    __shared__ float red[256];

    const int b = blockIdx.x;
    const int t = threadIdx.x;
    const int* mb = mask + (size_t)b * 65536;

    for (int idx = t; idx < 4096; idx += 256) {
        int i = idx >> 6, j = idx & 63;
        float v = (float)mb[(i * 4) * 256 + j * 4];
        ms[idx] = v;
        ms_g[(size_t)b * 4096 + idx] = v;
    }
    __syncthreads();

    float sms = 0.f, smbg = 0.f;
    for (int idx = t; idx < 4096; idx += 256) {
        int i = idx >> 6, j = idx & 63;
        float mx = 0.f;
        #pragma unroll
        for (int di = -1; di <= 1; ++di)
            #pragma unroll
            for (int dj = -1; dj <= 1; ++dj) {
                int ii = i + di, jj = j + dj;
                if (ii >= 0 && ii < 64 && jj >= 0 && jj < 64)
                    mx = fmaxf(mx, ms[ii * 64 + jj]);
            }
        float bg = fmaxf(mx - ms[idx], 0.f);
        mbg_g[(size_t)b * 4096 + idx] = bg;
        sms += ms[idx];
        smbg += bg;
    }
    red[t] = sms; __syncthreads();
    for (int s = 128; s > 0; s >>= 1) { if (t < s) red[t] += red[t + s]; __syncthreads(); }
    if (t == 0) sums_g[b] = red[0];
    __syncthreads();
    red[t] = smbg; __syncthreads();
    for (int s = 128; s > 0; s >>= 1) { if (t < s) red[t] += red[t + s]; __syncthreads(); }
    if (t == 0) sums_g[BATCH + b] = red[0];
    __syncthreads();

    for (int idx = t; idx < 4096; idx += 256) {
        int i = idx >> 6, j = idx & 63;
        float best = 1e30f;
        for (int ip = 0; ip < 64; ++ip) {
            float cand = fabsf((float)(i - ip)) + 128.f * (1.f - ms[ip * 64 + j]);
            best = fminf(best, cand);
        }
        gg[idx] = best;
    }
    __syncthreads();

    float lmax = 0.f;
    for (int idx = t; idx < 4096; idx += 256) {
        int i = idx >> 6, j = idx & 63;
        float best = 1e30f;
        for (int jp = 0; jp < 64; ++jp) {
            float gv = gg[i * 64 + jp];
            float dj = (float)(j - jp);
            best = fminf(best, gv * gv + dj * dj);
        }
        float d = sqrtf(best);
        dd[idx] = d;
        lmax = fmaxf(lmax, d);
    }
    red[t] = lmax; __syncthreads();
    for (int s = 128; s > 0; s >>= 1) { if (t < s) red[t] = fmaxf(red[t], red[t + s]); __syncthreads(); }
    float inv = 1.f / (red[0] + 1e-6f);
    for (int idx = t; idx < 4096; idx += 256)
        dist_g[(size_t)b * 4096 + idx] = dd[idx] * inv;
}

// ---------------------------------------------------------------------------
// Kernel 2: p_contrast[b,c]
// ---------------------------------------------------------------------------
__global__ __launch_bounds__(256) void pcon_kernel(
    const float* __restrict__ fs,
    const float* __restrict__ ms_g,
    const float* __restrict__ mbg_g,
    const float* __restrict__ sums_g,
    float* __restrict__ pcon_g)
{
    __shared__ float r1[256];
    __shared__ float r2[256];
    const int c = blockIdx.x, b = blockIdx.y, t = threadIdx.x;
    const float* f   = fs    + ((size_t)b * CCH + c) * HW;
    const float* msb = ms_g  + (size_t)b * HW;
    const float* bgb = mbg_g + (size_t)b * HW;
    float s1 = 0.f, s2 = 0.f;
    for (int idx = t; idx < HW; idx += 256) {
        float v = f[idx];
        s1 += v * msb[idx];
        s2 += v * bgb[idx];
    }
    r1[t] = s1; r2[t] = s2; __syncthreads();
    for (int s = 128; s > 0; s >>= 1) {
        if (t < s) { r1[t] += r1[t + s]; r2[t] += r2[t + s]; }
        __syncthreads();
    }
    if (t == 0)
        pcon_g[b * CCH + c] = r1[0] / (sums_g[b] + EPSV) - r2[0] / (sums_g[BATCH + b] + EPSV);
}

// ---------------------------------------------------------------------------
// Kernel 3: weight repack f32 -> f16
// ---------------------------------------------------------------------------
__global__ __launch_bounds__(256) void wconv_kernel(
    const float* __restrict__ gate_w,     // [512,512]
    const float* __restrict__ fusion_w,   // [512,513]
    _Float16* __restrict__ gwh,
    _Float16* __restrict__ fwh,
    float* __restrict__ fwlast)
{
    int idx = blockIdx.x * 256 + threadIdx.x;
    int o = idx >> 9, c = idx & 511;
    gwh[idx] = (_Float16)gate_w[idx];
    fwh[idx] = (_Float16)fusion_w[o * 513 + c];
    if (c == 0) fwlast[o] = fusion_w[o * 513 + 512];
}

// ---------------------------------------------------------------------------
// Kernel 4/5: WMMA GEMM  out[b,o,hw] = sum_c W[o,c] * X[b,c,hw]
//   block tile 128(M=o) x 256(N=hw) x 32(K=c); 8 wave32 waves in a 2x4 grid,
//   each wave owns 64x64 -> 4x4 v_wmma_f32_16x16x32_f16 accumulators.
//   A tile staged by TDM tensor_load_to_lds (hw-padded to pitch 40);
//   B tile staged/transposed by all threads; next B slab prefetched.
// ---------------------------------------------------------------------------
template <int MODE>
__global__ __launch_bounds__(256) void gemm_wmma(
    const _Float16* __restrict__ A_h,     // [512,512] f16 weights
    const float*    __restrict__ Bf,      // MODE0: f_s f32 [B,512,4096]
    const _Float16* __restrict__ Bh,      // MODE1: f_pur f16 [B,512,4096]
    const float*    __restrict__ fs,      // f_s (epilogue, MODE0)
    const float*    __restrict__ pcon,    // [B,512]
    const float*    __restrict__ gate_b,  // [512]
    const float*    __restrict__ fusion_b,// [512]
    const float*    __restrict__ fwlast,  // [512]
    const float*    __restrict__ dist,    // [B,4096]
    _Float16* __restrict__ fpur_out,      // MODE0 output
    float*    __restrict__ out)           // MODE1 output
{
    __shared__ _Float16 Alds[128 * 40];   // [row o][k], pitch 40 (TDM pad)
    __shared__ _Float16 Blds[256 * 40];   // [col hw][k], pitch 40

    const int b  = blockIdx.z;
    const int n0 = blockIdx.x * 256;      // hw tile
    const int o0 = blockIdx.y * 128;      // out-channel tile
    const int t    = threadIdx.x;
    const int lane = t & 31;
    const int w    = t >> 5;
    const int wm   = w >> 2;              // 2 M-slices of 64
    const int wn   = w & 3;               // 4 N-slices of 64
    const int lhi  = lane >> 4;
    const int lm   = lane & 15;

    v8f acc[4][4];
    const v8f vz = {0.f, 0.f, 0.f, 0.f, 0.f, 0.f, 0.f, 0.f};
    #pragma unroll
    for (int mi = 0; mi < 4; ++mi)
        #pragma unroll
        for (int ni = 0; ni < 4; ++ni) acc[mi][ni] = vz;

    for (int k0 = 0; k0 < CCH; k0 += 32) {
        // ---- A tile via Tensor Data Mover (wave 0 issues; EXEC-independent) ----
        if (w == 0) {
            tdm_load_a_tile(A_h + (size_t)o0 * CCH + k0, Alds);
            __builtin_amdgcn_s_wait_tensorcnt((short)0);
        }
        // ---- stage B transposed: Blds[col][k] ----
        if (MODE == 0) {
            #pragma unroll
            for (int it = 0; it < 8; ++it) {
                int lin = t + it * 256;            // 0..2047 quads
                int k   = lin >> 6;                // 0..31
                int col = (lin & 63) * 4;          // 0..252
                v4f v = *(const v4f*)(Bf + ((size_t)b * CCH + k0 + k) * HW + n0 + col);
                Blds[(col + 0) * 40 + k] = (_Float16)v[0];
                Blds[(col + 1) * 40 + k] = (_Float16)v[1];
                Blds[(col + 2) * 40 + k] = (_Float16)v[2];
                Blds[(col + 3) * 40 + k] = (_Float16)v[3];
            }
        } else {
            #pragma unroll
            for (int it = 0; it < 4; ++it) {
                int lin = t + it * 256;            // 0..1023 octets
                int k   = lin >> 5;                // 0..31
                int col = (lin & 31) * 8;          // 0..248
                v8h v = *(const v8h*)(Bh + ((size_t)b * CCH + k0 + k) * HW + n0 + col);
                #pragma unroll
                for (int i = 0; i < 8; ++i)
                    Blds[(col + i) * 40 + k] = v[i];
            }
        }
        // prefetch next K-slab of B into GL2 while we compute (global_prefetch_b8)
        if (k0 + 32 < CCH) {
            int kk = t & 31, cc = (t >> 5) * 32;
            if (MODE == 0)
                __builtin_prefetch(Bf + ((size_t)b * CCH + k0 + 32 + kk) * HW + n0 + cc, 0, 1);
            else
                __builtin_prefetch(Bh + ((size_t)b * CCH + k0 + 32 + kk) * HW + n0 + cc, 0, 1);
        }
        __syncthreads();

        // ---- fragments (documented wave32 VGPR layouts) ----
        // A 16x32 f16: lane<16 -> K {0..7,16..23}; lane>=16 -> K {8..15,24..31}
        v16h afr[4], bfr[4];
        const int klo = lhi * 8;
        #pragma unroll
        for (int mi = 0; mi < 4; ++mi) {
            int row = wm * 64 + mi * 16 + lm;
            v8h x0 = *(const v8h*)(Alds + row * 40 + klo);
            v8h x1 = *(const v8h*)(Alds + row * 40 + klo + 16);
            afr[mi] = __builtin_shufflevector(x0, x1,
                0, 1, 2, 3, 4, 5, 6, 7, 8, 9, 10, 11, 12, 13, 14, 15);
        }
        // B 32x16 f16: lane<16 -> K {0..7,16..23}; lane>=16 -> K {8..15,24..31}; N = lane&15
        #pragma unroll
        for (int ni = 0; ni < 4; ++ni) {
            int col = wn * 64 + ni * 16 + lm;
            v8h y0 = *(const v8h*)(Blds + col * 40 + lhi * 8);
            v8h y1 = *(const v8h*)(Blds + col * 40 + lhi * 8 + 16);
            bfr[ni] = __builtin_shufflevector(y0, y1,
                0, 1, 2, 3, 4, 5, 6, 7, 8, 9, 10, 11, 12, 13, 14, 15);
        }

        #pragma unroll
        for (int mi = 0; mi < 4; ++mi)
            #pragma unroll
            for (int ni = 0; ni < 4; ++ni)
                acc[mi][ni] = __builtin_amdgcn_wmma_f32_16x16x32_f16(
                    false, afr[mi], false, bfr[ni],
                    (short)0, acc[mi][ni], false, false);

        __syncthreads();
    }

    // ---- epilogue: C/D layout -> M = r + 8*(lane>>4), N = lane&15 ----
    #pragma unroll
    for (int mi = 0; mi < 4; ++mi) {
        #pragma unroll
        for (int ni = 0; ni < 4; ++ni) {
            v8f a = acc[mi][ni];
            #pragma unroll
            for (int r = 0; r < 8; ++r) {
                int o  = o0 + wm * 64 + mi * 16 + r + 8 * lhi;
                int hw = n0 + wn * 64 + ni * 16 + lm;
                float v = a[r];
                size_t oidx = ((size_t)b * CCH + o) * HW + hw;
                if (MODE == 0) {
                    float gt = 1.f / (1.f + expf(-(v + gate_b[o])));
                    float fp = fs[oidx] + gt * pcon[b * CCH + o];
                    fpur_out[oidx] = (_Float16)fp;
                } else {
                    float val = v + fusion_b[o] + fwlast[o] * dist[(size_t)b * HW + hw];
                    out[oidx] = fmaxf(val, 0.f);
                }
            }
        }
    }
}

// ---------------------------------------------------------------------------
extern "C" void kernel_launch(void* const* d_in, const int* in_sizes, int n_in,
                              void* d_out, int out_size, void* d_ws, size_t ws_size,
                              hipStream_t stream) {
    const float* fs       = (const float*)d_in[0];   // [8,512,64,64]
    const int*   smask    = (const int*)  d_in[1];   // [8,1,256,256]
    const float* gate_w   = (const float*)d_in[2];   // [512,512]
    const float* gate_b   = (const float*)d_in[3];   // [512]
    const float* fusion_w = (const float*)d_in[4];   // [512,513]
    const float* fusion_b = (const float*)d_in[5];   // [512]
    float* out = (float*)d_out;

    char* base = (char*)d_ws;
    size_t off = 0;
    auto carve = [&](size_t bytes) { size_t o = off; off = (off + bytes + 255) & ~(size_t)255; return o; };
    float*    ms_g    = (float*)   (base + carve((size_t)BATCH * HW * 4));
    float*    mbg_g   = (float*)   (base + carve((size_t)BATCH * HW * 4));
    float*    dist_g  = (float*)   (base + carve((size_t)BATCH * HW * 4));
    float*    sums_g  = (float*)   (base + carve(2 * BATCH * 4));
    float*    pcon_g  = (float*)   (base + carve((size_t)BATCH * CCH * 4));
    float*    fwlast  = (float*)   (base + carve(CCH * 4));
    _Float16* gwh     = (_Float16*)(base + carve((size_t)CCH * CCH * 2));
    _Float16* fwh     = (_Float16*)(base + carve((size_t)CCH * CCH * 2));
    _Float16* fpur_h  = (_Float16*)(base + carve((size_t)BATCH * CCH * HW * 2));
    (void)ws_size; (void)n_in; (void)in_sizes; (void)out_size;

    prep_kernel<<<BATCH, 256, 0, stream>>>(smask, ms_g, mbg_g, dist_g, sums_g);
    pcon_kernel<<<dim3(CCH, BATCH), 256, 0, stream>>>(fs, ms_g, mbg_g, sums_g, pcon_g);
    wconv_kernel<<<(CCH * CCH) / 256, 256, 0, stream>>>(gate_w, fusion_w, gwh, fwh, fwlast);
    gemm_wmma<0><<<dim3(HW / 256, CCH / 128, BATCH), 256, 0, stream>>>(
        gwh, fs, (const _Float16*)nullptr, fs, pcon_g, gate_b, fusion_b, fwlast, dist_g,
        fpur_h, out);
    gemm_wmma<1><<<dim3(HW / 256, CCH / 128, BATCH), 256, 0, stream>>>(
        fwh, (const float*)nullptr, fpur_h, fs, pcon_g, gate_b, fusion_b, fwlast, dist_g,
        fpur_h, out);
}
